// SwinDecoder_15985868276175
// MI455X (gfx1250) — compile-verified, matrix operand order checked by reading, not run on hardware
//
#include <hip/hip_runtime.h>
#include <hip/hip_fp16.h>
#include <hip/hip_bf16.h>

typedef __attribute__((ext_vector_type(16))) _Float16 v16h;
typedef __attribute__((ext_vector_type(8)))  float    v8f;

#define WSZ   7
#define NHD   6
#define BB    8
#define CIN   192
#define HIN   56
#define HH2   112
#define WW2   112
#define C2    96
#define LTOK  (HH2*WW2)           // 12544
#define TTOK  (BB*LTOK)           // 100352
#define HID   384
#define NWINX (WW2/WSZ)           // 16
#define GWIN  (BB*NWINX*NWINX)    // 2048
#define KCONV (CIN*9)             // 1728

// ---------------------------------------------------------------- WMMA helpers
// 16-bit A/B fragment layout (ISA 7.12.2): lane L holds row (L&15); K halves
// split by hi=L>>4 into two contiguous runs of 8 -> two 16B loads.
__device__ inline v16h ld_frag(const _Float16* base, int ld, int k0) {
  const int lane = threadIdx.x & 31;
  const _Float16* p = base + (lane & 15) * ld + k0 + (lane >> 4) * 8;
  v16h f;
  ((uint4*)&f)[0] = *(const uint4*)p;
  ((uint4*)&f)[1] = *(const uint4*)(p + 16);
  return f;
}

__device__ inline v8f wmma16(v16h a, v16h b, v8f c) {
  return __builtin_amdgcn_wmma_f32_16x16x32_f16(false, a, false, b, (short)0, c,
                                                false, false);
}

// ---------------------------------------------------------------- weight prep
__global__ __launch_bounds__(256) void cvt_kernel(const float* __restrict__ s,
                                                  _Float16* __restrict__ d, int n) {
  int i = blockIdx.x * 256 + threadIdx.x;
  if (i < n) d[i] = (_Float16)s[i];
}

// convBt[co][ (kh*3+kw)*192 + ci ] = up_w[co][ci][kh][kw]
__global__ __launch_bounds__(256) void convprep_kernel(const float* __restrict__ w,
                                                       _Float16* __restrict__ Bt) {
  int i = blockIdx.x * 256 + threadIdx.x;
  if (i >= C2 * KCONV) return;
  int co = i / KCONV, k = i % KCONV;
  int kk9 = k / CIN, ci = k % CIN;
  int kh = kk9 / 3, kw = kk9 % 3;
  Bt[i] = (_Float16)w[((co * CIN + ci) * 3 + kh) * 3 + kw];
}

// ------------------------------------------- upsample2x + conv3x3 + BN + ReLU
// Implicit GEMM: M = tokens (16/tile), N = 96 (6 waves x 16), K = 1728.
__global__ __launch_bounds__(192) void upconv_kernel(
    const float* __restrict__ x, const _Float16* __restrict__ wBt,
    const float* __restrict__ up_b, const float* __restrict__ bn_g,
    const float* __restrict__ bn_b, const float* __restrict__ bn_mean,
    const float* __restrict__ bn_var, float* __restrict__ X) {
  __shared__ _Float16 Alds[16 * 32];
  const int tok0 = blockIdx.x * 16;
  const int b    = tok0 / LTOK;
  const int hw0  = tok0 % LTOK;
  const int h    = hw0 / WW2;
  const int w0   = hw0 % WW2;          // 112 % 16 == 0: tile stays in one row
  const int lane = threadIdx.x & 31;
  const int wave = threadIdx.x >> 5;   // n-tile: channels wave*16..+15
  v8f acc = {};
  for (int c = 0; c < KCONV / 32; ++c) {
    const int kk9 = c / 6;             // kh*3+kw
    const int kh = kk9 / 3, kw = kk9 % 3;
    const int ci0 = (c % 6) * 32;
    __syncthreads();
    for (int e = threadIdx.x; e < 512; e += 192) {  // A tile [16][32]
      const int m = e & 15, kk = e >> 4;
      const int hh = h + kh - 1, ww = w0 + m + kw - 1;
      float v = 0.f;
      if (hh >= 0 && hh < HH2 && ww >= 0 && ww < WW2)
        v = x[((b * CIN + ci0 + kk) * HIN + (hh >> 1)) * HIN + (ww >> 1)];
      Alds[m * 32 + kk] = (_Float16)v;
    }
    __syncthreads();
    v16h af = ld_frag(Alds, 32, 0);
    v16h bf = ld_frag(wBt + (wave * 16) * KCONV, KCONV, c * 32);
    acc = wmma16(af, bf, acc);
  }
  const int co = wave * 16 + (lane & 15);
  const int hi = lane >> 4;
  const float bias = up_b[co];
  const float s  = bn_g[co] * rsqrtf(bn_var[co] + 1e-5f);
  const float mu = bn_mean[co], bb = bn_b[co];
#pragma unroll
  for (int r = 0; r < 8; ++r) {
    float v = (acc[r] + bias - mu) * s + bb;
    X[(size_t)(tok0 + r + hi * 8) * C2 + co] = fmaxf(v, 0.f);
  }
}

// ---------------------------------------------------------------- layer norm
__global__ __launch_bounds__(256) void ln_kernel(const float* __restrict__ X,
                                                 const float* __restrict__ g,
                                                 const float* __restrict__ bv,
                                                 _Float16* __restrict__ out) {
  const int wave = threadIdx.x >> 5, lane = threadIdx.x & 31;
  const int tok = blockIdx.x * 8 + wave;
  const float* xr = X + (size_t)tok * C2;
  float v0 = xr[lane], v1 = xr[lane + 32], v2 = xr[lane + 64];
  float s = v0 + v1 + v2;
#pragma unroll
  for (int m = 16; m >= 1; m >>= 1) s += __shfl_xor(s, m, 32);
  const float mean = s * (1.f / 96.f);
  float d0 = v0 - mean, d1 = v1 - mean, d2 = v2 - mean;
  float q = d0 * d0 + d1 * d1 + d2 * d2;
#pragma unroll
  for (int m = 16; m >= 1; m >>= 1) q += __shfl_xor(q, m, 32);
  const float inv = rsqrtf(q * (1.f / 96.f) + 1e-5f);
  _Float16* o = out + (size_t)tok * C2;
  o[lane]      = (_Float16)(d0 * inv * g[lane]      + bv[lane]);
  o[lane + 32] = (_Float16)(d1 * inv * g[lane + 32] + bv[lane + 32]);
  o[lane + 64] = (_Float16)(d2 * inv * g[lane + 64] + bv[lane + 64]);
}

// ------------------------------------------------------- generic WMMA GEMM
// C[M][N] = act(A[M][K] @ Bt[N][K]^T + bias) (+ residual). act: 0 none, 1 gelu.
// The 16 x K B-tile (shared by all 8 waves of the block) is staged in LDS once;
// next A chunk is prefetched (global_prefetch_b8) while WMMA executes.
__global__ __launch_bounds__(256) void gemm_kernel(
    const _Float16* __restrict__ A, const _Float16* __restrict__ Bt,
    const float* __restrict__ bias, int Kdim, int N,
    const float* __restrict__ residual, float* __restrict__ outF,
    _Float16* __restrict__ outH, int act) {
  __shared__ _Float16 Bl[16 * HID];                 // 12 KB max (Kdim <= 384)
  const int wave = threadIdx.x >> 5, lane = threadIdx.x & 31;
  const int m0 = (blockIdx.x * 8 + wave) * 16;
  const int n0 = blockIdx.y * 16;
  {
    const uint4* src = (const uint4*)(Bt + (size_t)n0 * Kdim);
    uint4* dst = (uint4*)Bl;
    const int nv = 16 * Kdim / 8;
    for (int i = threadIdx.x; i < nv; i += 256) dst[i] = src[i];
  }
  __syncthreads();
  v8f acc = {};
  const _Float16* Ab = A + (size_t)m0 * Kdim;
  const _Float16* Apf = Ab + (lane & 15) * Kdim + (lane >> 4) * 8;
  for (int k0 = 0; k0 < Kdim; k0 += 32) {
    if (k0 + 32 < Kdim) __builtin_prefetch(Apf + k0 + 32, 0, 0);
    acc = wmma16(ld_frag(Ab, Kdim, k0), ld_frag(Bl, Kdim, k0), acc);
  }
  const int col = n0 + (lane & 15);
  const int hi = lane >> 4;
  const float bcol = bias[col];
#pragma unroll
  for (int r = 0; r < 8; ++r) {
    const size_t row = m0 + r + hi * 8;
    float v = acc[r] + bcol;
    if (act == 1) v = 0.5f * v * (1.f + erff(v * 0.70710678f));
    if (residual) v += residual[row * N + col];
    if (outF) outF[row * N + col] = v;
    if (outH) outH[row * N + col] = (_Float16)v;
  }
}

// ---------------------------------------------- windowed attention (per head)
__device__ inline int swin_region(int c, int shift) {
  return (c < HH2 - WSZ) ? 0 : ((c < HH2 - shift) ? 1 : 2);
}

__global__ __launch_bounds__(32) void attn_kernel(
    const _Float16* __restrict__ QKV, const float* __restrict__ rel,
    _Float16* __restrict__ AT, int shift) {
  __shared__ _Float16 Qs[64 * 32];
  __shared__ _Float16 Ks[64 * 32];
  __shared__ _Float16 Vt[16 * 64];
  __shared__ float    Ss[64 * 64];
  __shared__ _Float16 Ps[64 * 64];
  const int g = blockIdx.x, head = blockIdx.y;
  const int b = g >> 8, win = g & 255;
  const int wh = win >> 4, ww = win & 15;
  const int lane = threadIdx.x;
  // gather Q (scaled), K, V^T with shift-roll mapping; pad to 64x32 / 16x64
  for (int m = lane; m < 64; m += 32) {
    long tok = -1;
    if (m < 49) {
      const int i = m / 7, j = m % 7;
      int hh = wh * WSZ + i + shift; if (hh >= HH2) hh -= HH2;
      int wc = ww * WSZ + j + shift; if (wc >= WW2) wc -= WW2;
      tok = (long)b * LTOK + hh * WW2 + wc;
    }
    for (int d = 0; d < 16; ++d) {
      _Float16 qv = (_Float16)0, kv = (_Float16)0, vv = (_Float16)0;
      if (tok >= 0) {
        const _Float16* p = QKV + (size_t)tok * (3 * C2) + head * 16 + d;
        qv = (_Float16)((float)p[0] * 0.25f);   // hd^-0.5 = 1/4
        kv = p[C2];
        vv = p[2 * C2];
      }
      Qs[m * 32 + d] = qv; Qs[m * 32 + 16 + d] = (_Float16)0;
      Ks[m * 32 + d] = kv; Ks[m * 32 + 16 + d] = (_Float16)0;
      Vt[d * 64 + m] = vv;
    }
  }
  __syncthreads();
  // S = Q @ K^T + rel-bias (+ shifted-window mask)
  for (int mt = 0; mt < 4; ++mt)
    for (int nt = 0; nt < 4; ++nt) {
      v8f c = {};
      c = wmma16(ld_frag(Qs + mt * 16 * 32, 32, 0),
                 ld_frag(Ks + nt * 16 * 32, 32, 0), c);
      const int colm = nt * 16 + (lane & 15);   // key token
      const int hi = lane >> 4;
      if (colm < 49) {
        const int i2 = colm / 7, j2 = colm % 7;
        const int r2 = shift ? swin_region(wh * WSZ + i2, shift) * 3 +
                               swin_region(ww * WSZ + j2, shift) : 0;
#pragma unroll
        for (int r = 0; r < 8; ++r) {
          const int rown = mt * 16 + r + hi * 8;  // query token
          float v = c[r];
          if (rown < 49) {
            const int i1 = rown / 7, j1 = rown % 7;
            v += rel[((i1 - i2 + 6) * 13 + (j1 - j2 + 6)) * NHD + head];
            if (shift) {
              const int r1 = swin_region(wh * WSZ + i1, shift) * 3 +
                             swin_region(ww * WSZ + j1, shift);
              if (r1 != r2) v -= 100.f;
            }
          }
          Ss[rown * 64 + colm] = v;
        }
      }
    }
  __syncthreads();
  // softmax over the 49 valid keys; write P (f16), zero-pad
  for (int rr = 0; rr < 2; ++rr) {
    const int n = lane + rr * 32;
    if (n < 49) {
      float mx = -1e30f;
      for (int m = 0; m < 49; ++m) mx = fmaxf(mx, Ss[n * 64 + m]);
      float sum = 0.f;
      for (int m = 0; m < 49; ++m) sum += __expf(Ss[n * 64 + m] - mx);
      const float inv = 1.f / sum;
      for (int m = 0; m < 49; ++m)
        Ps[n * 64 + m] = (_Float16)(__expf(Ss[n * 64 + m] - mx) * inv);
      for (int m = 49; m < 64; ++m) Ps[n * 64 + m] = (_Float16)0;
    } else {
      for (int m = 0; m < 64; ++m) Ps[n * 64 + m] = (_Float16)0;
    }
  }
  __syncthreads();
  // O = P @ V  -> scatter back through roll mapping
  for (int mt = 0; mt < 4; ++mt) {
    v8f c = {};
    for (int kc = 0; kc < 2; ++kc)
      c = wmma16(ld_frag(Ps + mt * 16 * 64, 64, kc * 32),
                 ld_frag(Vt, 64, kc * 32), c);
    const int d = lane & 15, hi = lane >> 4;
#pragma unroll
    for (int r = 0; r < 8; ++r) {
      const int n = mt * 16 + r + hi * 8;
      if (n < 49) {
        const int i = n / 7, j = n % 7;
        int hh = wh * WSZ + i + shift; if (hh >= HH2) hh -= HH2;
        int wc = ww * WSZ + j + shift; if (wc >= WW2) wc -= WW2;
        AT[((size_t)b * LTOK + hh * WW2 + wc) * C2 + head * 16 + d] =
            (_Float16)c[r];
      }
    }
  }
}

// -------------------------------------------- 1x1 conv + ReLU, NCHW scatter
__global__ __launch_bounds__(256) void gemm_out_kernel(
    const _Float16* __restrict__ A, const _Float16* __restrict__ Bt,
    const float* __restrict__ bias, float* __restrict__ out) {
  const int wave = threadIdx.x >> 5, lane = threadIdx.x & 31;
  const int m0 = (blockIdx.x * 8 + wave) * 16;
  const int n0 = blockIdx.y * 16;
  v8f acc = {};
  const _Float16* Ab = A + (size_t)m0 * C2;
  const _Float16* Bb = Bt + (size_t)n0 * C2;
  for (int k0 = 0; k0 < C2; k0 += 32)
    acc = wmma16(ld_frag(Ab, C2, k0), ld_frag(Bb, C2, k0), acc);
  const int col = n0 + (lane & 15);
  const int hi = lane >> 4;
  const float bcol = bias[col];
#pragma unroll
  for (int r = 0; r < 8; ++r) {
    const int row = m0 + r + hi * 8;
    const int b = row / LTOK, hw = row % LTOK;
    out[((size_t)b * 48 + col) * LTOK + hw] = fmaxf(acc[r] + bcol, 0.f);
  }
}

// ---------------------------------------------------------------- launcher
extern "C" void kernel_launch(void* const* d_in, const int* in_sizes, int n_in,
                              void* d_out, int out_size, void* d_ws, size_t ws_size,
                              hipStream_t stream) {
  const float* x      = (const float*)d_in[0];
  const float* up_w   = (const float*)d_in[1];
  const float* up_b   = (const float*)d_in[2];
  const float* bn_g   = (const float*)d_in[3];
  const float* bn_b   = (const float*)d_in[4];
  const float* bn_mean= (const float*)d_in[5];
  const float* bn_var = (const float*)d_in[6];
  const float* n1g    = (const float*)d_in[7];
  const float* n1b    = (const float*)d_in[8];
  const float* qkv_w  = (const float*)d_in[9];
  const float* qkv_b  = (const float*)d_in[10];
  const float* proj_w = (const float*)d_in[11];
  const float* proj_b = (const float*)d_in[12];
  const float* rel    = (const float*)d_in[13];
  const float* n2g    = (const float*)d_in[14];
  const float* n2b    = (const float*)d_in[15];
  const float* f1w    = (const float*)d_in[16];
  const float* f1b    = (const float*)d_in[17];
  const float* f2w    = (const float*)d_in[18];
  const float* f2b    = (const float*)d_in[19];
  const float* out_w  = (const float*)d_in[20];
  const float* out_b  = (const float*)d_in[21];
  float* out = (float*)d_out;

  char* wp = (char*)d_ws;
  auto carve = [&](size_t bytes) -> void* {
    void* p = (void*)wp;
    wp += (bytes + 255) & ~(size_t)255;
    return p;
  };
  float*    X    = (float*)carve((size_t)TTOK * C2 * 4);    // residual stream
  _Float16* XN   = (_Float16*)carve((size_t)TTOK * C2 * 2); // LN output
  _Float16* SCR  = (_Float16*)carve((size_t)TTOK * HID * 2);// QKV (288) / MLP hidden (384)
  _Float16* ATb  = (_Float16*)carve((size_t)TTOK * C2 * 2); // attention out
  _Float16* XH   = (_Float16*)carve((size_t)TTOK * C2 * 2); // final f16 stream
  _Float16* convBt = (_Float16*)carve((size_t)C2 * KCONV * 2);
  _Float16* qkvBt  = (_Float16*)carve((size_t)2 * 3 * C2 * C2 * 2);
  _Float16* projBt = (_Float16*)carve((size_t)2 * C2 * C2 * 2);
  _Float16* f1Bt   = (_Float16*)carve((size_t)2 * HID * C2 * 2);
  _Float16* f2Bt   = (_Float16*)carve((size_t)2 * C2 * HID * 2);
  _Float16* outBt  = (_Float16*)carve((size_t)48 * C2 * 2);

  auto cvt = [&](const float* s, _Float16* d, int n) {
    cvt_kernel<<<(n + 255) / 256, 256, 0, stream>>>(s, d, n);
  };
  cvt(qkv_w, qkvBt, 2 * 3 * C2 * C2);
  cvt(proj_w, projBt, 2 * C2 * C2);
  cvt(f1w, f1Bt, 2 * HID * C2);
  cvt(f2w, f2Bt, 2 * C2 * HID);
  cvt(out_w, outBt, 48 * C2);
  convprep_kernel<<<(C2 * KCONV + 255) / 256, 256, 0, stream>>>(up_w, convBt);

  upconv_kernel<<<TTOK / 16, 192, 0, stream>>>(x, convBt, up_b, bn_g, bn_b,
                                               bn_mean, bn_var, X);

  const dim3 gm(TTOK / 128, 1);
  for (int i = 0; i < 2; ++i) {
    const int shift = i ? (WSZ / 2) : 0;
    ln_kernel<<<TTOK / 8, 256, 0, stream>>>(X, n1g + i * C2, n1b + i * C2, XN);
    gemm_kernel<<<dim3(gm.x, 3 * C2 / 16), 256, 0, stream>>>(
        XN, qkvBt + (size_t)i * 3 * C2 * C2, qkv_b + i * 3 * C2, C2, 3 * C2,
        nullptr, nullptr, SCR, 0);
    attn_kernel<<<dim3(GWIN, NHD), 32, 0, stream>>>(SCR, rel + i * 169 * NHD,
                                                    ATb, shift);
    gemm_kernel<<<dim3(gm.x, C2 / 16), 256, 0, stream>>>(
        ATb, projBt + (size_t)i * C2 * C2, proj_b + i * C2, C2, C2,
        X, X, nullptr, 0);
    ln_kernel<<<TTOK / 8, 256, 0, stream>>>(X, n2g + i * C2, n2b + i * C2, XN);
    gemm_kernel<<<dim3(gm.x, HID / 16), 256, 0, stream>>>(
        XN, f1Bt + (size_t)i * HID * C2, f1b + i * HID, C2, HID,
        nullptr, nullptr, SCR, 1);
    gemm_kernel<<<dim3(gm.x, C2 / 16), 256, 0, stream>>>(
        SCR, f2Bt + (size_t)i * C2 * HID, f2b + i * C2, HID, C2,
        X, X, XH, 0);
  }
  gemm_out_kernel<<<dim3(gm.x, 3), 256, 0, stream>>>(XH, outBt, out_b, out);
  (void)in_sizes; (void)n_in; (void)out_size; (void)ws_size;
}
